// GraphSAGELayer_37254546326038
// MI455X (gfx1250) — matrix operand order repaired and checked by reading, not compiled, not run on hardware
//
#include <hip/hip_runtime.h>
#include <hip/hip_bf16.h>

// Problem constants (match reference)
#define BB 16
#define NN 16384
#define DD 128
#define EE 65536
#define LN_EPS 1e-5f
#define WPITCH 136   // LDS row pitch in halfs: 272B = 68 dwords -> bank-conflict-free B frags

typedef __attribute__((ext_vector_type(16))) _Float16 v16h;
typedef __attribute__((ext_vector_type(8)))  _Float16 v8h;
typedef __attribute__((ext_vector_type(8)))  float    v8f;

// ---------------------------------------------------------------------------
// Kernel 0: zero the aggregation buffer (aliased with d_out) and degree buffer
// ---------------------------------------------------------------------------
__global__ __launch_bounds__(256) void sage_zero_kernel(float* __restrict__ agg,
                                                        float* __restrict__ deg) {
  size_t i = (size_t)blockIdx.x * blockDim.x + threadIdx.x;
  const size_t nagg = (size_t)BB * NN * DD;
  if (i < nagg) {
    agg[i] = 0.0f;
  } else {
    size_t j = i - nagg;
    if (j < (size_t)BB * NN) deg[j] = 0.0f;
  }
}

// ---------------------------------------------------------------------------
// Kernel 1: edge scatter-sum. One wave per (batch, edge): 32 lanes cover the
// 128 features as float4 chunks; src/dst loads are wave-uniform.
// ---------------------------------------------------------------------------
__global__ __launch_bounds__(256) void sage_scatter_kernel(
    const float* __restrict__ x, const long long* __restrict__ ei,
    float* __restrict__ agg, float* __restrict__ deg) {
  size_t t = (size_t)blockIdx.x * blockDim.x + threadIdx.x;
  size_t edge = t >> 5;               // global (b,e) edge id
  int chunk = (int)(t & 31);          // 4 floats per lane
  size_t b = edge / EE;
  size_t e = edge - b * EE;
  size_t src = (size_t)ei[e]       + b * NN;
  size_t dst = (size_t)ei[EE + e]  + b * NN;
  const float4 v = *(const float4*)(x + src * DD + chunk * 4);
  float* p = agg + dst * DD + chunk * 4;
  atomicAdd(p + 0, v.x);
  atomicAdd(p + 1, v.y);
  atomicAdd(p + 2, v.z);
  atomicAdd(p + 3, v.w);
  if (chunk == 0) atomicAdd(deg + dst, 1.0f);
}

// ---------------------------------------------------------------------------
// Helper: read one 32x16 f16 B fragment from pitched LDS.
// Lane layout: N = lane&15, K = kb..kb+7 then kb+16..kb+23, kb = (lane>>4)*8.
// Pitch 272B is only 16B-aligned -> load as two v8h and merge.
// ---------------------------------------------------------------------------
__device__ __forceinline__ v16h load_b_frag(const _Float16* __restrict__ lw,
                                            int nt, int nl, int kw, int kb) {
  const _Float16* p = lw + (size_t)(nt * 16 + nl) * WPITCH + kw * 32 + kb;
  v8h lo = *(const v8h*)(p);
  v8h hi = *(const v8h*)(p + 16);
  return __builtin_shufflevector(lo, hi, 0, 1, 2, 3, 4, 5, 6, 7,
                                 8, 9, 10, 11, 12, 13, 14, 15);
}

// ---------------------------------------------------------------------------
// Kernel 2: fused dual-GEMM (WMMA f16 -> f32 acc) + bias + LayerNorm + ReLU.
// One wave per 16-node tile; 8 waves per block. Weights are staged (f32->f16)
// into pitched LDS once per block and shared by all 8 waves. agg is read from
// `out` and the same 16 rows are overwritten with the final result.
// ---------------------------------------------------------------------------
__global__ __launch_bounds__(256) void sage_gemm_ln_kernel(
    const float* __restrict__ x, const float* __restrict__ deg,
    const float* __restrict__ wself, const float* __restrict__ wneigh,
    const float* __restrict__ bias, const float* __restrict__ gamma,
    const float* __restrict__ beta, float* __restrict__ out) {
  extern __shared__ _Float16 smem[];
  _Float16* lws = smem;                       // [DD][WPITCH]
  _Float16* lwn = smem + DD * WPITCH;         // [DD][WPITCH]

  // ---- Stage weights into LDS (f32 -> f16, pitched rows) ----
  {
    int t = threadIdx.x;                      // 0..255
    int row = t >> 1;                         // 0..127
    int col0 = (t & 1) * 64;                  // 0 or 64
    const float* gs = wself  + row * DD + col0;
    const float* gn = wneigh + row * DD + col0;
    _Float16* ps = lws + row * WPITCH + col0;
    _Float16* pn = lwn + row * WPITCH + col0;
#pragma unroll
    for (int i = 0; i < 64; ++i) {
      ps[i] = (_Float16)gs[i];
      pn[i] = (_Float16)gn[i];
    }
  }
  __syncthreads();

  const int lane = threadIdx.x & 31;
  const int wave = threadIdx.x >> 5;
  const int tile = blockIdx.x * 8 + wave;     // 16-node tile index
  const int nl   = lane & 15;                 // lane-in-half
  const int half = lane >> 4;                 // 0 or 1
  const int ka   = half * 8;                  // A-frag K base
  const int kb   = half * 8;                  // B-frag K base

  // This lane's A-matrix row (M = nl) for both x and neigh fragments
  const size_t anode = (size_t)tile * 16 + nl;
  const float* xrow = x   + anode * DD;
  const float* arow = out + anode * DD;       // agg aliased in out
  const float dg    = deg[anode];
  // fast v_rcp_f32: feeds an f16-quantized fragment, 1-ulp rcp is plenty
  const float scale = __builtin_amdgcn_rcpf(fmaxf(dg, 1.0f));

  // Build A fragments for all 4 K-windows (ISA 16-bit A 16x32 layout:
  // elements 0..7 = K[ka..ka+7], elements 8..15 = K[ka+16..ka+23]).
  v16h ax[4], an[4];
#pragma unroll
  for (int kw = 0; kw < 4; ++kw) {
    const float* px = xrow + kw * 32 + ka;
    const float* pa = arow + kw * 32 + ka;
    v16h a1, a2;
#pragma unroll
    for (int i = 0; i < 8; ++i) {
      a1[i]     = (_Float16)px[i];
      a1[8 + i] = (_Float16)px[16 + i];
      a2[i]     = (_Float16)(pa[i] * scale);
      a2[8 + i] = (_Float16)(pa[16 + i] * scale);
    }
    ax[kw] = a1;
    an[kw] = a2;
  }

  // Dual GEMM: 8 n-tiles x 4 k-windows x 2 matrices = 64 WMMAs per wave,
  // B fragments served from LDS.
  v8f acc[8];
#pragma unroll
  for (int nt = 0; nt < 8; ++nt) {
    v8f c = {};
#pragma unroll
    for (int kw = 0; kw < 4; ++kw) {
      v16h bs = load_b_frag(lws, nt, nl, kw, kb);
      c = __builtin_amdgcn_wmma_f32_16x16x32_f16(
          false, ax[kw], false, bs, (short)0, c, false, false);
      v16h bn = load_b_frag(lwn, nt, nl, kw, kb);
      c = __builtin_amdgcn_wmma_f32_16x16x32_f16(
          false, an[kw], false, bn, (short)0, c, false, false);
    }
    acc[nt] = c;
  }

  // Epilogue: bias + LayerNorm over the 128 features + ReLU.
  // C/D layout: VGPR r holds M = r + 8*half, N = nl + 16*nt.
  float bn_[8], gn_[8], btn_[8];
#pragma unroll
  for (int nt = 0; nt < 8; ++nt) {
    int n = nt * 16 + nl;
    bn_[nt]  = bias[n];
    gn_[nt]  = gamma[n];
    btn_[nt] = beta[n];
  }
#pragma unroll
  for (int nt = 0; nt < 8; ++nt)
#pragma unroll
    for (int r = 0; r < 8; ++r) acc[nt][r] += bn_[nt];

#pragma unroll
  for (int r = 0; r < 8; ++r) {
    float s = 0.0f, q = 0.0f;
#pragma unroll
    for (int nt = 0; nt < 8; ++nt) {
      float v = acc[nt][r];
      s += v;
      q += v * v;
    }
    // Reduce across the 16 lanes of this half (masks 1,2,4,8 stay in-half).
#pragma unroll
    for (int m = 1; m <= 8; m <<= 1) {
      s += __shfl_xor(s, m, 32);
      q += __shfl_xor(q, m, 32);
    }
    const float mu   = s * (1.0f / 128.0f);
    const float var  = q * (1.0f / 128.0f) - mu * mu;
    const float rstd = __builtin_amdgcn_rsqf(var + LN_EPS);  // fast v_rsq_f32

    float* orow = out + ((size_t)tile * 16 + r + half * 8) * DD;
#pragma unroll
    for (int nt = 0; nt < 8; ++nt) {
      float v = (acc[nt][r] - mu) * rstd * gn_[nt] + btn_[nt];
      orow[nt * 16 + nl] = fmaxf(v, 0.0f);
    }
  }
}

// ---------------------------------------------------------------------------
extern "C" void kernel_launch(void* const* d_in, const int* in_sizes, int n_in,
                              void* d_out, int out_size, void* d_ws, size_t ws_size,
                              hipStream_t stream) {
  (void)in_sizes; (void)n_in; (void)out_size; (void)ws_size;
  const float*     x      = (const float*)d_in[0];
  const long long* ei     = (const long long*)d_in[1];   // int64 [2,E]
  const float*     wself  = (const float*)d_in[3];
  const float*     wneigh = (const float*)d_in[4];
  const float*     bias   = (const float*)d_in[5];
  const float*     gamma  = (const float*)d_in[6];
  const float*     beta   = (const float*)d_in[7];
  float* out = (float*)d_out;

  // Workspace layout: deg (B*N f32 = 1MB)
  float* deg = (float*)d_ws;

  // 0) zero agg (aliased in d_out) + deg
  size_t nzero = (size_t)BB * NN * DD + (size_t)BB * NN;
  sage_zero_kernel<<<(unsigned)((nzero + 255) / 256), 256, 0, stream>>>(out, deg);

  // 1) edge scatter-sum + degree (1M edges, 32 lanes each)
  size_t nsc = (size_t)BB * EE * 32;
  sage_scatter_kernel<<<(unsigned)(nsc / 256), 256, 0, stream>>>(x, ei, out, deg);

  // 2) fused dual-GEMM (WMMA, LDS-staged weights) + bias + LayerNorm + ReLU
  const int ntiles = BB * NN / 16;             // 16384 tiles, 8 per block
  const size_t lds_bytes = (size_t)2 * DD * WPITCH * sizeof(_Float16);  // 69632
  sage_gemm_ln_kernel<<<ntiles / 8, 256, lds_bytes, stream>>>(
      x, deg, wself, wneigh, bias, gamma, beta, out);
}
